// GraphAttentionLayer_66967130079332
// MI455X (gfx1250) — matrix operand order repaired and checked by reading, not compiled, not run on hardware
//
#include <hip/hip_runtime.h>
#include <hip/hip_bf16.h>

// ---------- types for WMMA ----------
typedef __attribute__((ext_vector_type(16))) __bf16        v16bf;
typedef __attribute__((ext_vector_type(8)))  float         v8f;
typedef __attribute__((ext_vector_type(8)))  unsigned int  v8u;
typedef __attribute__((ext_vector_type(4)))  unsigned int  v4u;
typedef __attribute__((ext_vector_type(4)))  int           v4i;
typedef __attribute__((ext_vector_type(4)))  float         v4f;

#define NEG_INF_V (-9.0e15f)
#define LRELU     (0.01f)
#define PSTRIDE   36     // padded LDS row stride (dwords) to avoid bank conflicts

// round-to-nearest-even f32 -> bf16
__device__ __forceinline__ unsigned short f2bf(float f) {
    unsigned int u = __builtin_bit_cast(unsigned int, f);
    u += 0x7fffu + ((u >> 16) & 1u);
    return (unsigned short)(u >> 16);
}

union Frag {
    struct { v4u lo, hi; } p;
    v8u   u;
    v16bf v;
};

// ---------- fused-DPP all-reduces (no LDS traffic) ----------
__device__ __forceinline__ float red16_sum(float x) {
    asm("v_add_f32_dpp %0, %1, %0 quad_perm:[1,0,3,2] row_mask:0xf bank_mask:0xf bound_ctrl:1" : "+v"(x) : "v"(x));
    asm("v_add_f32_dpp %0, %1, %0 quad_perm:[2,3,0,1] row_mask:0xf bank_mask:0xf bound_ctrl:1" : "+v"(x) : "v"(x));
    asm("v_add_f32_dpp %0, %1, %0 row_half_mirror row_mask:0xf bank_mask:0xf bound_ctrl:1" : "+v"(x) : "v"(x));
    asm("v_add_f32_dpp %0, %1, %0 row_mirror row_mask:0xf bank_mask:0xf bound_ctrl:1" : "+v"(x) : "v"(x));
    return x;
}
// 8-lane all-reduce: rows owned by 8 contiguous, 8-aligned lanes
__device__ __forceinline__ float red8_max(float x) {
    asm("v_max_num_f32_dpp %0, %1, %0 quad_perm:[1,0,3,2] row_mask:0xf bank_mask:0xf bound_ctrl:1" : "+v"(x) : "v"(x));
    asm("v_max_num_f32_dpp %0, %1, %0 quad_perm:[2,3,0,1] row_mask:0xf bank_mask:0xf bound_ctrl:1" : "+v"(x) : "v"(x));
    asm("v_max_num_f32_dpp %0, %1, %0 row_half_mirror row_mask:0xf bank_mask:0xf bound_ctrl:1" : "+v"(x) : "v"(x));
    return x;
}
__device__ __forceinline__ float red8_sum(float x) {
    asm("v_add_f32_dpp %0, %1, %0 quad_perm:[1,0,3,2] row_mask:0xf bank_mask:0xf bound_ctrl:1" : "+v"(x) : "v"(x));
    asm("v_add_f32_dpp %0, %1, %0 quad_perm:[2,3,0,1] row_mask:0xf bank_mask:0xf bound_ctrl:1" : "+v"(x) : "v"(x));
    asm("v_add_f32_dpp %0, %1, %0 row_half_mirror row_mask:0xf bank_mask:0xf bound_ctrl:1" : "+v"(x) : "v"(x));
    return x;
}

// =====================================================================
// Phase 1: h = x @ W (f32 compute).  Per block: 16 rows of x.
// =====================================================================
__global__ __launch_bounds__(256) void gat_phase1(
    const float* __restrict__ x, const float* __restrict__ W,
    const float* __restrict__ a, unsigned short* __restrict__ hT,
    float* __restrict__ s1, float* __restrict__ s2)
{
    __shared__ __align__(16) float          xs[16 * 256];
    __shared__ __align__(16) unsigned short htile[16 * 128];

    const int i0 = blockIdx.x * 16;
    const int t  = threadIdx.x;

    const float* xrow = x + (size_t)i0 * 256;
    for (int k = t; k < 16 * 256; k += 256) xs[k] = xrow[k];
    __syncthreads();

    const int r  = t >> 4;
    const int q  = t & 15;
    const int n0 = q * 8;

    float acc[8] = {0.f, 0.f, 0.f, 0.f, 0.f, 0.f, 0.f, 0.f};
    for (int k = 0; k < 256; ++k) {
        const float  xv = xs[r * 256 + k];
        const float* wr = W + k * 128 + n0;
        #pragma unroll
        for (int j = 0; j < 8; ++j) acc[j] += xv * wr[j];
    }

    float p1 = 0.f, p2 = 0.f;
    #pragma unroll
    for (int j = 0; j < 8; ++j) {
        p1 += acc[j] * a[n0 + j];
        p2 += acc[j] * a[128 + n0 + j];
    }
    p1 = red16_sum(p1);
    p2 = red16_sum(p2);
    if (q == 0) { s1[i0 + r] = p1; s2[i0 + r] = p2; }

    #pragma unroll
    for (int j = 0; j < 8; ++j) htile[r * 128 + n0 + j] = f2bf(acc[j]);
    __syncthreads();

    const int n  = t >> 1;
    const int ic = (t & 1) * 8;
    union { unsigned short s[8]; v4u v; } pk;
    #pragma unroll
    for (int j = 0; j < 8; ++j) pk.s[j] = htile[(ic + j) * 128 + n];
    *(v4u*)(hT + (size_t)n * 8192 + i0 + ic) = pk.v;
}

// =====================================================================
// Phase 2: fused masked-softmax attention, online softmax.
// Block = 32 output rows, 256 threads = 8 waves.  64-column chunks:
// each wave owns a 32x16 C tile (2 accumulators) and reuses each B
// fragment for two WMMAs -> hT/L2 traffic halved.  adj streamed once.
// =====================================================================
__global__ __launch_bounds__(256) void gat_phase2(
    const int* __restrict__ adj, const unsigned short* __restrict__ hT,
    const float* __restrict__ s1, const float* __restrict__ s2,
    float* __restrict__ out)
{
    __shared__ __align__(16) unsigned int lds_p[2][32 * PSTRIDE]; // bf16 pairs, 32x64
    __shared__ __align__(16) float lds_scale[2][32];
    __shared__ float lds_l[32];

    const int t  = threadIdx.x;
    const int i0 = blockIdx.x * 32;

    // ---- producer-role mapping: 32 rows x 64 cols / 256 threads ----
    const int r = t >> 3;                 // local row 0..31
    const int q = t & 7;                  // 8 columns per thread
    const int* adjrow = adj + (size_t)(i0 + r) * 8192;
    const float s1r = s1[i0 + r];

    // ---- consumer-role mapping (per wave) ----
    const int lane    = t & 31;
    const int w       = t >> 5;
    const int n       = lane & 15;
    const int hiHalf  = lane >> 4;
    const int kbase   = hiHalf * 8;
    const int koff    = hiHalf * 16;
    const unsigned short* hcol = hT + (size_t)(w * 16 + n) * 8192;

    float m = -3.0e38f;
    float l = 0.f;
    v8f acc0 = {0.f, 0.f, 0.f, 0.f, 0.f, 0.f, 0.f, 0.f};   // rows 0..15
    v8f acc1 = {0.f, 0.f, 0.f, 0.f, 0.f, 0.f, 0.f, 0.f};   // rows 16..31

    // ---- chunk body: consume pipe regs for chunk c (64 cols) ----
    auto chunk = [&](int c, bool reload,
                     v4i& xa0, v4i& xa1, v4f& xu0, v4f& xu1,
                     Frag& xBL, Frag& xBH) {
        const int buf = c & 1;
        const int jj  = c * 64 + q * 8;

        // producer math on 8 pipelined columns
        float e[8], vv[8];
        bool  b[8];
        #pragma unroll
        for (int k = 0; k < 4; ++k) {
            float ea = s1r + xu0[k];
            float eb = s1r + xu1[k];
            ea = ea > 0.f ? ea : LRELU * ea;
            eb = eb > 0.f ? eb : LRELU * eb;
            e[k] = ea;       e[4 + k] = eb;
            b[k] = xa0[k] > 0;   b[4 + k] = xa1[k] > 0;
            vv[k]     = b[k]     ? ea : NEG_INF_V;
            vv[4 + k] = b[4 + k] ? eb : NEG_INF_V;
        }

        // reload adj/s2 for chunk c+2 (main loop: always in-bounds)
        if (reload) {
            const int jjn = jj + 128;
            xa0 = *(const v4i*)(adjrow + jjn);
            xa1 = *(const v4i*)(adjrow + jjn + 4);
            xu0 = *(const v4f*)(s2 + jjn);
            xu1 = *(const v4f*)(s2 + jjn + 4);
            __builtin_prefetch(adjrow + jj + 256, 0, 3);  // 4 chunks ahead
        }

        // online softmax (8-lane fused-DPP all-reduces)
        float cmax = fmaxf(fmaxf(fmaxf(vv[0], vv[1]), fmaxf(vv[2], vv[3])),
                           fmaxf(fmaxf(vv[4], vv[5]), fmaxf(vv[6], vv[7])));
        cmax = red8_max(cmax);
        const float mn = fmaxf(m, cmax);
        const float sc = __expf(m - mn);
        float p[8], ps = 0.f;
        #pragma unroll
        for (int k = 0; k < 8; ++k) {
            p[k] = b[k] ? __expf(e[k] - mn) : 0.f;
            ps += p[k];
        }
        ps = red8_sum(ps);
        l = l * sc + ps;
        m = mn;

        v4u pk;
        #pragma unroll
        for (int k = 0; k < 4; ++k)
            pk[k] = (unsigned int)f2bf(p[2 * k]) |
                    ((unsigned int)f2bf(p[2 * k + 1]) << 16);
        *(v4u*)&lds_p[buf][r * PSTRIDE + q * 4] = pk;
        if (q == 0) lds_scale[buf][r] = sc;
        __syncthreads();

        // consumer: rescale both accumulators, 4 WMMAs reusing 2 B frags
        #pragma unroll
        for (int v = 0; v < 8; ++v) {
            acc0[v] *= lds_scale[buf][v + kbase];
            acc1[v] *= lds_scale[buf][16 + v + kbase];
        }

        Frag AL0, AL1;
        AL0.p.lo = *(const v4u*)&lds_p[buf][n * PSTRIDE + hiHalf * 8];
        AL0.p.hi = *(const v4u*)&lds_p[buf][n * PSTRIDE + hiHalf * 8 + 4];
        AL1.p.lo = *(const v4u*)&lds_p[buf][(n + 16) * PSTRIDE + hiHalf * 8];
        AL1.p.hi = *(const v4u*)&lds_p[buf][(n + 16) * PSTRIDE + hiHalf * 8 + 4];
        acc0 = __builtin_amdgcn_wmma_f32_16x16x32_bf16(
                   false, AL0.v, false, xBL.v, (short)0, acc0, false, false);
        acc1 = __builtin_amdgcn_wmma_f32_16x16x32_bf16(
                   false, AL1.v, false, xBL.v, (short)0, acc1, false, false);

        Frag AH0, AH1;
        AH0.p.lo = *(const v4u*)&lds_p[buf][n * PSTRIDE + 16 + hiHalf * 8];
        AH0.p.hi = *(const v4u*)&lds_p[buf][n * PSTRIDE + 16 + hiHalf * 8 + 4];
        AH1.p.lo = *(const v4u*)&lds_p[buf][(n + 16) * PSTRIDE + 16 + hiHalf * 8];
        AH1.p.hi = *(const v4u*)&lds_p[buf][(n + 16) * PSTRIDE + 16 + hiHalf * 8 + 4];
        acc0 = __builtin_amdgcn_wmma_f32_16x16x32_bf16(
                   false, AH0.v, false, xBH.v, (short)0, acc0, false, false);
        acc1 = __builtin_amdgcn_wmma_f32_16x16x32_bf16(
                   false, AH1.v, false, xBH.v, (short)0, acc1, false, false);

        // reload B fragments for chunk c+2 after all WMMAs consumed them
        if (reload) {
            const v4u* nb0 = (const v4u*)(hcol + (c + 2) * 64 + koff);
            xBL.p.lo = nb0[0]; xBL.p.hi = nb0[1];
            const v4u* nb1 = (const v4u*)(hcol + (c + 2) * 64 + 32 + koff);
            xBH.p.lo = nb1[0]; xBH.p.hi = nb1[1];
        }
    };

    // ---- pipeline prologue: preload chunks 0 and 1 ----
    v4i a00 = *(const v4i*)(adjrow + q * 8);
    v4i a01 = *(const v4i*)(adjrow + q * 8 + 4);
    v4f u00 = *(const v4f*)(s2 + q * 8);
    v4f u01 = *(const v4f*)(s2 + q * 8 + 4);
    v4i a10 = *(const v4i*)(adjrow + 64 + q * 8);
    v4i a11 = *(const v4i*)(adjrow + 64 + q * 8 + 4);
    v4f u10 = *(const v4f*)(s2 + 64 + q * 8);
    v4f u11 = *(const v4f*)(s2 + 64 + q * 8 + 4);
    Frag B0L, B0H, B1L, B1H;
    {
        const v4u* p0 = (const v4u*)(hcol + koff);
        B0L.p.lo = p0[0]; B0L.p.hi = p0[1];
        const v4u* p1 = (const v4u*)(hcol + 32 + koff);
        B0H.p.lo = p1[0]; B0H.p.hi = p1[1];
        const v4u* p2 = (const v4u*)(hcol + 64 + koff);
        B1L.p.lo = p2[0]; B1L.p.hi = p2[1];
        const v4u* p3 = (const v4u*)(hcol + 96 + koff);
        B1H.p.lo = p3[0]; B1H.p.hi = p3[1];
    }

    // main loop: chunks 0..125, reloads c+2 (always in-bounds)
    for (int c = 0; c < 126; c += 2) {
        chunk(c,     true, a00, a01, u00, u01, B0L, B0H);
        chunk(c + 1, true, a10, a11, u10, u11, B1L, B1H);
    }
    // peeled tail: chunks 126, 127 (no reload)
    chunk(126, false, a00, a01, u00, u01, B0L, B0H);
    chunk(127, false, a10, a11, u10, u11, B1L, B1H);

    if (q == 0) lds_l[r] = l;
    __syncthreads();

    #pragma unroll
    for (int v = 0; v < 8; ++v) {
        const int row0 = v + kbase;            // C layout: M = v + 8*hiHalf
        const int row1 = 16 + v + kbase;
        float d0 = lds_l[row0]; d0 = d0 > 1e-30f ? d0 : 1e-30f;
        float d1 = lds_l[row1]; d1 = d1 > 1e-30f ? d1 : 1e-30f;
        float y0 = acc0[v] / d0;
        float y1 = acc1[v] / d1;
        y0 = y0 > 0.f ? y0 : (__expf(y0) - 1.f);   // ELU
        y1 = y1 > 0.f ? y1 : (__expf(y1) - 1.f);
        out[(size_t)(i0 + row0) * 128 + w * 16 + n] = y0;
        out[(size_t)(i0 + row1) * 128 + w * 16 + n] = y1;
    }
}

// =====================================================================
extern "C" void kernel_launch(void* const* d_in, const int* in_sizes, int n_in,
                              void* d_out, int out_size, void* d_ws, size_t ws_size,
                              hipStream_t stream) {
    const float* x   = (const float*)d_in[0];   // 8192 x 256
    const int*   adj = (const int*)  d_in[1];   // 8192 x 8192
    const float* W   = (const float*)d_in[2];   // 256 x 128
    const float* a   = (const float*)d_in[3];   // 256 x 1
    float*       out = (float*)d_out;           // 8192 x 128

    char* ws = (char*)d_ws;
    unsigned short* hT = (unsigned short*)ws;                 // 128*8192 bf16 = 2 MB
    float* s1 = (float*)(ws + (size_t)128 * 8192 * 2);        // 8192 f32
    float* s2 = s1 + 8192;                                    // 8192 f32

    gat_phase1<<<512, 256, 0, stream>>>(x, W, a, hT, s1, s2);
    gat_phase2<<<256, 256, 0, stream>>>(adj, hT, s1, s2, out);
}
